// MyModel_89206470738237
// MI455X (gfx1250) — matrix operand order repaired and checked by reading, not compiled, not run on hardware
//
#include <hip/hip_runtime.h>
#include <math.h>

// ---------------------------------------------------------------------------
// Problem constants (from reference)
// ---------------------------------------------------------------------------
#define NN      50000
#define EE      800000
#define TT      12
#define HOR     12
#define F_IN    2
#define EMB     16
#define HID     64
#define DPAD    128          // padded K dimension for all GEMMs
#define ROWT    (NN / 16)    // 3125 row tiles of 16 (N divisible by 16)

typedef __attribute__((ext_vector_type(16))) __bf16 v16bf;
typedef __attribute__((ext_vector_type(8)))  float  v8f;

union ABfrag { v16bf v; unsigned int u[8]; uint4 q[2]; };

// ---------------------------------------------------------------------------
// bf16 helpers
// ---------------------------------------------------------------------------
__device__ __forceinline__ float bf2f(unsigned short h) {
    unsigned int u = ((unsigned int)h) << 16;
    float f; __builtin_memcpy(&f, &u, 4); return f;
}
__device__ __forceinline__ unsigned short f2bf(float f) {
    unsigned int u; __builtin_memcpy(&u, &f, 4);
    u += 0x7FFFu + ((u >> 16) & 1u);           // round-to-nearest-even
    return (unsigned short)(u >> 16);
}
__device__ __forceinline__ unsigned int pack2bf(float a, float b) {
    return (unsigned int)f2bf(a) | ((unsigned int)f2bf(b) << 16);
}

// fast sigmoid / tanh built on v_exp_f32 + v_rcp_f32
__device__ __forceinline__ float fast_sigmoid(float x) {
    return __builtin_amdgcn_rcpf(1.f + __expf(-x));
}
__device__ __forceinline__ float fast_tanh(float x) {
    // tanh(x) = 2/(1+exp(-2x)) - 1
    return fmaf(2.f, __builtin_amdgcn_rcpf(1.f + __expf(-2.f * x)), -1.f);
}

// Packed bf16 atomic add (GLOBAL_ATOMIC_PK_ADD_BF16, no return -> STOREcnt).
__device__ __forceinline__ void atomic_pk_add_bf16(unsigned short* addr, unsigned int val) {
    asm volatile("global_atomic_pk_add_bf16 %0, %1, off" :: "v"(addr), "v"(val) : "memory");
}

// ---------------------------------------------------------------------------
// Degree precompute; coef(e) = a[src]*b[dst] factorizes, so we only keep
// per-node a = rsqrt(deg_out), b = rsqrt(deg_in)  (reused 8x per timestep).
// ---------------------------------------------------------------------------
__global__ void deg_kernel(const int* __restrict__ src, const int* __restrict__ dst,
                           unsigned int* __restrict__ dout, unsigned int* __restrict__ din) {
    int e = blockIdx.x * blockDim.x + threadIdx.x;
    if (e >= EE) return;
    atomicAdd(&dout[src[e]], 1u);
    atomicAdd(&din[dst[e]], 1u);
}

__global__ void rsq_kernel(const unsigned int* __restrict__ degO, const unsigned int* __restrict__ degI,
                           float* __restrict__ rsqo, float* __restrict__ rsqi) {
    int n = blockIdx.x * blockDim.x + threadIdx.x;
    if (n >= NN) return;
    unsigned int a = degO[n]; if (a < 1u) a = 1u;
    unsigned int b = degI[n]; if (b < 1u) b = 1u;
    rsqo[n] = rsqrtf((float)a);
    rsqi[n] = rsqrtf((float)b);
}

// ---------------------------------------------------------------------------
// Weight pack: f32 [D, OutW] -> bf16 WMMA-B-fragment order, zero rows past D.
// Layout: uint index = ((ct*4 + kk)*32 + lane)*8 + v
//   col = ct*16 + lane%16;  k = kk*32 + (v>=4 ? 16 : 0) + (lane>=16 ? 8 : 0) + 2(v&3)  (pair k,k+1)
// ---------------------------------------------------------------------------
__global__ void pack_weight(const float* __restrict__ src, int D, int OutW,
                            unsigned int* __restrict__ dst) {
    int tid = blockIdx.x * blockDim.x + threadIdx.x;
    int total = (OutW / 16) * 4 * 32 * 8;
    if (tid >= total) return;
    int v    = tid & 7;
    int lane = (tid >> 3) & 31;
    int kk   = (tid >> 8) & 3;
    int ct   = tid >> 10;
    int col  = ct * 16 + (lane & 15);
    int k    = kk * 32 + ((v >> 2) ? 16 : 0) + ((lane >> 4) ? 8 : 0) + 2 * (v & 3);
    float lo = (k     < D) ? src[(size_t)k       * OutW + col] : 0.f;
    float hi = (k + 1 < D) ? src[(size_t)(k + 1) * OutW + col] : 0.f;
    dst[tid] = pack2bf(lo, hi);
}

// ---------------------------------------------------------------------------
// zcat builders: per node write bf16 [N,128]
//   zcat : unscaled concat (GEMM A-matrix / gprop self term)     (always)
//   zs   : zcat * a[node]  (scatter source)                      (full=1 only)
//   acc  : zeroed accumulator for the edge scatter               (full=1 only)
// One thread per (node, 8-column chunk) => coalesced uint4 stores.
// ---------------------------------------------------------------------------
__device__ __forceinline__ void store_built(int node, int c8, const float* vals, float a,
                                            int full, unsigned short* zcat,
                                            unsigned short* zs, unsigned short* acc) {
    uint4 p;
    p.x = pack2bf(vals[0], vals[1]);         p.y = pack2bf(vals[2], vals[3]);
    p.z = pack2bf(vals[4], vals[5]);         p.w = pack2bf(vals[6], vals[7]);
    ((uint4*)zcat)[(size_t)node * 16 + c8] = p;
    if (full) {
        uint4 q;
        q.x = pack2bf(vals[0]*a, vals[1]*a); q.y = pack2bf(vals[2]*a, vals[3]*a);
        q.z = pack2bf(vals[4]*a, vals[5]*a); q.w = pack2bf(vals[6]*a, vals[7]*a);
        ((uint4*)zs )[(size_t)node * 16 + c8] = q;
        ((uint4*)acc)[(size_t)node * 16 + c8] = make_uint4(0u, 0u, 0u, 0u);
    }
}

__global__ void build_enc0(const float* __restrict__ x_t, const float* __restrict__ Wf,
                           const float* __restrict__ bf, const float* __restrict__ h,
                           const float* __restrict__ rg, int use_r,
                           const float* __restrict__ a_t, int full,
                           unsigned short* __restrict__ zcat,
                           unsigned short* __restrict__ zs,
                           unsigned short* __restrict__ acc) {
    int tid = blockIdx.x * blockDim.x + threadIdx.x;
    int node = tid >> 4, c8 = tid & 15;
    if (node >= NN) return;
    float x0 = x_t[node * 2 + 0], x1 = x_t[node * 2 + 1];
    float vals[8];
    int j0 = c8 * 8;
    #pragma unroll
    for (int i = 0; i < 8; ++i) {
        int j = j0 + i; float v;
        if (j < EMB) {
            v = x0 * Wf[j] + x1 * Wf[EMB + j] + bf[j];
        } else if (j < EMB + HID) {
            int hj = j - EMB;
            v = h[(size_t)node * HID + hj];
            if (use_r) v *= rg[(size_t)node * DPAD + hj];
        } else v = 0.f;
        vals[i] = v;
    }
    store_built(node, c8, vals, a_t[node], full, zcat, zs, acc);
}

__global__ void build_l1(const float* __restrict__ hin, const float* __restrict__ h,
                         const float* __restrict__ rg, int use_r,
                         const float* __restrict__ a_t, int full,
                         unsigned short* __restrict__ zcat,
                         unsigned short* __restrict__ zs,
                         unsigned short* __restrict__ acc) {
    int tid = blockIdx.x * blockDim.x + threadIdx.x;
    int node = tid >> 4, c8 = tid & 15;
    if (node >= NN) return;
    float vals[8];
    int j0 = c8 * 8;
    #pragma unroll
    for (int i = 0; i < 8; ++i) {
        int j = j0 + i; float v;
        if (j < HID) {
            v = hin[(size_t)node * HID + j];
        } else {
            int hj = j - HID;
            v = h[(size_t)node * HID + hj];
            if (use_r) v *= rg[(size_t)node * DPAD + hj];
        }
        vals[i] = v;
    }
    store_built(node, c8, vals, a_t[node], full, zcat, zs, acc);
}

__global__ void build_dec0(const float* __restrict__ y, const float* __restrict__ h,
                           const float* __restrict__ rg, int use_r,
                           const float* __restrict__ a_t, int full,
                           unsigned short* __restrict__ zcat,
                           unsigned short* __restrict__ zs,
                           unsigned short* __restrict__ acc) {
    int tid = blockIdx.x * blockDim.x + threadIdx.x;
    int node = tid >> 4, c8 = tid & 15;
    if (node >= NN) return;
    float vals[8];
    int j0 = c8 * 8;
    #pragma unroll
    for (int i = 0; i < 8; ++i) {
        int j = j0 + i; float v;
        if (j == 0) {
            v = y[node];
        } else if (j < 1 + HID) {
            int hj = j - 1;
            v = h[(size_t)node * HID + hj];
            if (use_r) v *= rg[(size_t)node * DPAD + hj];
        } else v = 0.f;
        vals[i] = v;
    }
    store_built(node, c8, vals, a_t[node], full, zcat, zs, acc);
}

// ---------------------------------------------------------------------------
// Edge scatter (hot loop): acc[dst] += zs[src].  Pure copy-atomic.
// NC = number of 16-bf16 (32B) chunks actually scattered (skip zero pad);
// STRIDE = row stride in elements.  8 packed-bf16 atomics per thread.
// ---------------------------------------------------------------------------
template<int NC, int STRIDE>
__global__ void scatter_kernel(const int* __restrict__ src, const int* __restrict__ dst,
                               const unsigned short* __restrict__ zs,
                               unsigned short* __restrict__ acc) {
    int tid = blockIdx.x * blockDim.x + threadIdx.x;
    int e = tid / NC, c = tid % NC;
    if (e >= EE) return;
    int s = src[e], d = dst[e];
    const uint4* zrow = (const uint4*)(zs + (size_t)s * STRIDE);
    uint4 z0 = zrow[c * 2 + 0];
    uint4 z1 = zrow[c * 2 + 1];
    unsigned short* base = acc + (size_t)d * STRIDE + c * 16;
    atomic_pk_add_bf16(base +  0, z0.x);
    atomic_pk_add_bf16(base +  2, z0.y);
    atomic_pk_add_bf16(base +  4, z0.z);
    atomic_pk_add_bf16(base +  6, z0.w);
    atomic_pk_add_bf16(base +  8, z1.x);
    atomic_pk_add_bf16(base + 10, z1.y);
    atomic_pk_add_bf16(base + 12, z1.z);
    atomic_pk_add_bf16(base + 14, z1.w);
}

// ---------------------------------------------------------------------------
// Combine (gate path): gp[n] = zcat[n] + b[n] * acc[n]
// ---------------------------------------------------------------------------
__device__ __forceinline__ unsigned int combine2(unsigned int z, unsigned int a, float b) {
    float lo = fmaf(b, bf2f((unsigned short)(a & 0xFFFFu)), bf2f((unsigned short)(z & 0xFFFFu)));
    float hi = fmaf(b, bf2f((unsigned short)(a >> 16)),     bf2f((unsigned short)(z >> 16)));
    return pack2bf(lo, hi);
}

__global__ void combine_kernel(const unsigned short* __restrict__ zcat,
                               const unsigned short* __restrict__ acc,
                               const float* __restrict__ b_t,
                               unsigned short* __restrict__ gp) {
    int tid = blockIdx.x * blockDim.x + threadIdx.x;
    int node = tid >> 4, c8 = tid & 15;
    if (node >= NN) return;
    float b = b_t[node];
    uint4 z = ((const uint4*)zcat)[(size_t)node * 16 + c8];
    uint4 a = ((const uint4*)acc )[(size_t)node * 16 + c8];
    uint4 o;
    o.x = combine2(z.x, a.x, b);
    o.y = combine2(z.y, a.y, b);
    o.z = combine2(z.z, a.z, b);
    o.w = combine2(z.w, a.w, b);
    ((uint4*)gp)[(size_t)node * 16 + c8] = o;
}

// ---------------------------------------------------------------------------
// Candidate path (GEMM commuted before gprop; 64-wide scatter):
//   m = zcat @ Wc (f32, no bias) -> ms16 = bf16(m*a), acc64 = 0
//   scatter: acc64[dst] += ms16[src]
//   update:  c = tanh(m + b*acc64 + bc);  h = u*h + (1-u)*c
// ---------------------------------------------------------------------------
__global__ void build64(const float* __restrict__ m, const float* __restrict__ a_t,
                        unsigned short* __restrict__ ms16, unsigned short* __restrict__ acc64) {
    int tid = blockIdx.x * blockDim.x + threadIdx.x;
    int node = tid >> 3, c8 = tid & 7;
    if (node >= NN) return;
    float a = a_t[node];
    const float4* mr = (const float4*)(m + (size_t)node * HID);
    float4 f0 = mr[c8 * 2 + 0];
    float4 f1 = mr[c8 * 2 + 1];
    uint4 q;
    q.x = pack2bf(f0.x * a, f0.y * a); q.y = pack2bf(f0.z * a, f0.w * a);
    q.z = pack2bf(f1.x * a, f1.y * a); q.w = pack2bf(f1.z * a, f1.w * a);
    ((uint4*)ms16 )[(size_t)node * 8 + c8] = q;
    ((uint4*)acc64)[(size_t)node * 8 + c8] = make_uint4(0u, 0u, 0u, 0u);
}

__global__ void update64(const float* __restrict__ m, const unsigned short* __restrict__ acc64,
                         const float* __restrict__ b_t, const float* __restrict__ bias,
                         const float* __restrict__ gates, float* __restrict__ h) {
    int tid = blockIdx.x * blockDim.x + threadIdx.x;
    int node = tid >> 3, c8 = tid & 7;
    if (node >= NN) return;
    float b = b_t[node];
    const float4* mr = (const float4*)(m + (size_t)node * HID);
    float4 f0 = mr[c8 * 2 + 0];
    float4 f1 = mr[c8 * 2 + 1];
    float mv[8] = { f0.x, f0.y, f0.z, f0.w, f1.x, f1.y, f1.z, f1.w };
    uint4 aq = ((const uint4*)acc64)[(size_t)node * 8 + c8];
    unsigned int au[4] = { aq.x, aq.y, aq.z, aq.w };
    #pragma unroll
    for (int i = 0; i < 8; ++i) {
        int j = c8 * 8 + i;
        unsigned int w = au[i >> 1];
        float av = bf2f((unsigned short)((i & 1) ? (w >> 16) : (w & 0xFFFFu)));
        float c = fast_tanh(fmaf(b, av, mv[i]) + bias[j]);
        float u = gates[(size_t)node * DPAD + HID + j];
        float ho = h[(size_t)node * HID + j];
        h[(size_t)node * HID + j] = u * ho + (1.f - u) * c;
    }
}

// ---------------------------------------------------------------------------
// WMMA fragment loads (two b128 loads per fragment; layouts per ISA 7.12.2)
// ---------------------------------------------------------------------------
__device__ __forceinline__ void load_A_frag(const uint4* A4row, int kk, int hi16, ABfrag& a) {
    a.q[0] = A4row[kk * 4 + hi16];
    a.q[1] = A4row[kk * 4 + hi16 + 2];
}
__device__ __forceinline__ void load_B_frag(const uint4* B4, int ct, int kk, int lane, ABfrag& b) {
    int base = ((ct * 4 + kk) * 32 + lane) * 2;
    b.q[0] = B4[base];
    b.q[1] = B4[base + 1];
}

// gates = sigmoid(gp @ Wg + bg); one wave per 16x16 tile, 8 waves = 8 col tiles.
__global__ void __launch_bounds__(256) gemm_gates(const unsigned short* __restrict__ A,
                                                  const unsigned int* __restrict__ Wpk,
                                                  const float* __restrict__ bias,
                                                  float* __restrict__ gates) {
    int wave = threadIdx.x >> 5;
    int lane = threadIdx.x & 31;
    int hi16 = lane >> 4;
    int l16  = lane & 15;
    int row0 = blockIdx.x * 16;            // exactly ROWT blocks
    int ct   = wave;                       // 0..7
    int col  = ct * 16 + l16;
    const uint4* A4row = (const uint4*)(A + (size_t)(row0 + l16) * DPAD);
    const uint4* B4    = (const uint4*)Wpk;

    v8f acc = {};
    #pragma unroll
    for (int kk = 0; kk < 4; ++kk) {
        ABfrag a, b;
        load_A_frag(A4row, kk, hi16, a);
        load_B_frag(B4, ct, kk, lane, b);
        acc = __builtin_amdgcn_wmma_f32_16x16x32_bf16(false, a.v, false, b.v,
                                                      (short)0, acc, false, false);
    }
    float bs = bias[col];
    #pragma unroll
    for (int r = 0; r < 8; ++r) {
        int row = row0 + r + (hi16 ? 8 : 0);
        gates[(size_t)row * DPAD + col] = fast_sigmoid(acc[r] + bs);
    }
}

// m = zcat @ Wc (f32 out, NO bias/activation -- gprop still to be applied).
__global__ void __launch_bounds__(256) gemm_cand_pre(const unsigned short* __restrict__ A,
                                                     const unsigned int* __restrict__ Wpk,
                                                     float* __restrict__ m) {
    int wave = threadIdx.x >> 5;
    int lane = threadIdx.x & 31;
    int hi16 = lane >> 4;
    int l16  = lane & 15;
    int rowTile = blockIdx.x * 2 + (wave >> 2);
    int ct      = wave & 3;
    if (rowTile >= ROWT) return;           // wave-uniform exit keeps EXEC full
    int row0 = rowTile * 16;
    int col  = ct * 16 + l16;              // 0..63
    const uint4* A4row = (const uint4*)(A + (size_t)(row0 + l16) * DPAD);
    const uint4* B4    = (const uint4*)Wpk;

    v8f acc = {};
    #pragma unroll
    for (int kk = 0; kk < 4; ++kk) {
        ABfrag a, b;
        load_A_frag(A4row, kk, hi16, a);
        load_B_frag(B4, ct, kk, lane, b);
        acc = __builtin_amdgcn_wmma_f32_16x16x32_bf16(false, a.v, false, b.v,
                                                      (short)0, acc, false, false);
    }
    #pragma unroll
    for (int r = 0; r < 8; ++r) {
        int row = row0 + r + (hi16 ? 8 : 0);
        m[(size_t)row * HID + col] = acc[r];
    }
}

// ---------------------------------------------------------------------------
// Decoder output projection: y = h1 @ Wp + bp  (Wp [64,1])
// ---------------------------------------------------------------------------
__global__ void proj_kernel(const float* __restrict__ h1, const float* __restrict__ Wp,
                            const float* __restrict__ bp, float* __restrict__ y,
                            float* __restrict__ out_t) {
    int node = blockIdx.x * blockDim.x + threadIdx.x;
    if (node >= NN) return;
    float s = bp[0];
    #pragma unroll
    for (int j = 0; j < HID; ++j) s += h1[(size_t)node * HID + j] * Wp[j];
    y[node] = s;
    out_t[node] = s;
}

// ---------------------------------------------------------------------------
// Host orchestration
// ---------------------------------------------------------------------------
extern "C" void kernel_launch(void* const* d_in, const int* in_sizes, int n_in,
                              void* d_out, int out_size, void* d_ws, size_t ws_size,
                              hipStream_t stream) {
    (void)in_sizes; (void)n_in; (void)out_size; (void)ws_size;

    const int*   edges  = (const int*)  d_in[0];   // [T,2,E]
    const float* x      = (const float*)d_in[1];   // [T,N,2]
    const float* W_feat = (const float*)d_in[2];
    const float* b_feat = (const float*)d_in[3];
    const float* eWg0 = (const float*)d_in[4];  const float* ebg0 = (const float*)d_in[5];
    const float* eWc0 = (const float*)d_in[6];  const float* ebc0 = (const float*)d_in[7];
    const float* eWg1 = (const float*)d_in[8];  const float* ebg1 = (const float*)d_in[9];
    const float* eWc1 = (const float*)d_in[10]; const float* ebc1 = (const float*)d_in[11];
    const float* dWg0 = (const float*)d_in[12]; const float* dbg0 = (const float*)d_in[13];
    const float* dWc0 = (const float*)d_in[14]; const float* dbc0 = (const float*)d_in[15];
    const float* dWg1 = (const float*)d_in[16]; const float* dbg1 = (const float*)d_in[17];
    const float* dWc1 = (const float*)d_in[18]; const float* dbc1 = (const float*)d_in[19];
    const float* dWp  = (const float*)d_in[20]; const float* dbp  = (const float*)d_in[21];
    float* out = (float*)d_out;                    // [HOR,N,1]

    // workspace carve-out (~130 MB total)
    char* ws = (char*)d_ws; size_t off = 0;
    auto alloc = [&](size_t bytes) -> void* {
        void* p = ws + off; off += (bytes + 255) & ~(size_t)255; return p;
    };
    float*          rsqo  = (float*)alloc((size_t)TT * NN * 4);
    float*          rsqi  = (float*)alloc((size_t)TT * NN * 4);
    unsigned short* zcat  = (unsigned short*)alloc((size_t)NN * DPAD * 2);
    unsigned short* zs    = (unsigned short*)alloc((size_t)NN * DPAD * 2);
    unsigned short* acc   = (unsigned short*)alloc((size_t)NN * DPAD * 2);
    unsigned short* gp    = (unsigned short*)alloc((size_t)NN * DPAD * 2);
    float*          gates = (float*)alloc((size_t)NN * DPAD * 4);
    float*          m     = (float*)alloc((size_t)NN * HID * 4);
    unsigned short* ms16  = (unsigned short*)alloc((size_t)NN * HID * 2);
    unsigned short* acc64 = (unsigned short*)alloc((size_t)NN * HID * 2);
    float*          h0    = (float*)alloc((size_t)NN * HID * 4);
    float*          h1    = (float*)alloc((size_t)NN * HID * 4);
    float*          ybuf  = (float*)alloc((size_t)NN * 4);
    unsigned int*   degO  = (unsigned int*)alloc((size_t)NN * 4);
    unsigned int*   degI  = (unsigned int*)alloc((size_t)NN * 4);
    const int PKG = 8 * 4 * 32 * 8;   // packed uints for OutW=128 (32 KB)
    const int PKC = 4 * 4 * 32 * 8;   // packed uints for OutW=64  (16 KB)
    unsigned int* eWg0p = (unsigned int*)alloc(PKG * 4);
    unsigned int* eWc0p = (unsigned int*)alloc(PKC * 4);
    unsigned int* eWg1p = (unsigned int*)alloc(PKG * 4);
    unsigned int* eWc1p = (unsigned int*)alloc(PKC * 4);
    unsigned int* dWg0p = (unsigned int*)alloc(PKG * 4);
    unsigned int* dWc0p = (unsigned int*)alloc(PKC * 4);
    unsigned int* dWg1p = (unsigned int*)alloc(PKG * 4);
    unsigned int* dWc1p = (unsigned int*)alloc(PKC * 4);

    const int TPB = 256;
    const int edgeBlocks  = (EE + TPB - 1) / TPB;
    const int buildBlocks = (NN * 16) / TPB;                 // N*16 threads (exact)
    const int b64Blocks   = (NN * 8 + TPB - 1) / TPB;        // N*8 threads
    const int nodeBlocks  = (NN + TPB - 1) / TPB;
    const int candBlocks  = (ROWT + 1) / 2;

    // zero-init recurrent state each call (graph replays must be reproducible)
    hipMemsetAsync(h0,   0, (size_t)NN * HID * 4, stream);
    hipMemsetAsync(h1,   0, (size_t)NN * HID * 4, stream);
    hipMemsetAsync(ybuf, 0, (size_t)NN * 4,       stream);

    // pack weights into WMMA-B fragment order (bf16) once per launch
    {
        const int pkgBlocks = (PKG + TPB - 1) / TPB;
        const int pkcBlocks = (PKC + TPB - 1) / TPB;
        pack_weight<<<pkgBlocks, TPB, 0, stream>>>(eWg0, EMB + HID, DPAD, eWg0p);
        pack_weight<<<pkcBlocks, TPB, 0, stream>>>(eWc0, EMB + HID, HID,  eWc0p);
        pack_weight<<<pkgBlocks, TPB, 0, stream>>>(eWg1, 2 * HID,   DPAD, eWg1p);
        pack_weight<<<pkcBlocks, TPB, 0, stream>>>(eWc1, 2 * HID,   HID,  eWc1p);
        pack_weight<<<pkgBlocks, TPB, 0, stream>>>(dWg0, 1 + HID,   DPAD, dWg0p);
        pack_weight<<<pkcBlocks, TPB, 0, stream>>>(dWc0, 1 + HID,   HID,  dWc0p);
        pack_weight<<<pkgBlocks, TPB, 0, stream>>>(dWg1, 2 * HID,   DPAD, dWg1p);
        pack_weight<<<pkcBlocks, TPB, 0, stream>>>(dWc1, 2 * HID,   HID,  dWc1p);
    }

    // per-timestep per-node normalization factors (reused 8x each)
    for (int t = 0; t < TT; ++t) {
        const int* src = edges + (size_t)t * 2 * EE;
        const int* dst = src + EE;
        hipMemsetAsync(degO, 0, (size_t)NN * 4, stream);
        hipMemsetAsync(degI, 0, (size_t)NN * 4, stream);
        deg_kernel<<<edgeBlocks, TPB, 0, stream>>>(src, dst, degO, degI);
        rsq_kernel<<<nodeBlocks, TPB, 0, stream>>>(degO, degI,
                                                   rsqo + (size_t)t * NN, rsqi + (size_t)t * NN);
    }

    // gates tail: scatter (NC chunks of real width) -> combine -> WMMA+sigmoid
    auto gates_seq = [&](const int* src, const int* dst, const float* bt,
                         const unsigned int* Wgp, const float* bgp, int nc5) {
        if (nc5) scatter_kernel<5, 128><<<(EE * 5 + TPB - 1) / TPB, TPB, 0, stream>>>(src, dst, zs, acc);
        else     scatter_kernel<8, 128><<<(EE * 8 + TPB - 1) / TPB, TPB, 0, stream>>>(src, dst, zs, acc);
        combine_kernel<<<buildBlocks, TPB, 0, stream>>>(zcat, acc, bt, gp);
        gemm_gates<<<ROWT, TPB, 0, stream>>>(gp, Wgp, bgp, gates);
    };
    // cand tail: WMMA first (64-wide), then 64-wide gprop + fused GRU update
    auto cand_seq = [&](const int* src, const int* dst, const float* at, const float* bt,
                        const unsigned int* Wcp, const float* bcp, float* hbuf) {
        gemm_cand_pre<<<candBlocks, TPB, 0, stream>>>(zcat, Wcp, m);
        build64<<<b64Blocks, TPB, 0, stream>>>(m, at, ms16, acc64);
        scatter_kernel<4, 64><<<(EE * 4 + TPB - 1) / TPB, TPB, 0, stream>>>(src, dst, ms16, acc64);
        update64<<<b64Blocks, TPB, 0, stream>>>(m, acc64, bt, bcp, gates, hbuf);
    };

    // -------------------- encoder --------------------
    for (int t = 0; t < TT; ++t) {
        const int* src = edges + (size_t)t * 2 * EE;
        const int* dst = src + EE;
        const float* at = rsqo + (size_t)t * NN;
        const float* bt = rsqi + (size_t)t * NN;
        const float* x_t = x + (size_t)t * NN * F_IN;

        // layer 0 (real width 80 -> 5 chunks)
        build_enc0<<<buildBlocks, TPB, 0, stream>>>(x_t, W_feat, b_feat, h0, nullptr, 0, at, 1, zcat, zs, acc);
        gates_seq(src, dst, bt, eWg0p, ebg0, 1);
        build_enc0<<<buildBlocks, TPB, 0, stream>>>(x_t, W_feat, b_feat, h0, gates, 1, at, 0, zcat, zs, acc);
        cand_seq(src, dst, at, bt, eWc0p, ebc0, h0);

        // layer 1 (width 128 -> 8 chunks)
        build_l1<<<buildBlocks, TPB, 0, stream>>>(h0, h1, nullptr, 0, at, 1, zcat, zs, acc);
        gates_seq(src, dst, bt, eWg1p, ebg1, 0);
        build_l1<<<buildBlocks, TPB, 0, stream>>>(h0, h1, gates, 1, at, 0, zcat, zs, acc);
        cand_seq(src, dst, at, bt, eWc1p, ebc1, h1);
    }

    // -------------------- decoder --------------------
    for (int t = 0; t < HOR; ++t) {
        const int* src = edges + (size_t)t * 2 * EE;
        const int* dst = src + EE;
        const float* at = rsqo + (size_t)t * NN;
        const float* bt = rsqi + (size_t)t * NN;

        // layer 0 (real width 65 -> 5 chunks covers 80 cols incl. zero pad)
        build_dec0<<<buildBlocks, TPB, 0, stream>>>(ybuf, h0, nullptr, 0, at, 1, zcat, zs, acc);
        gates_seq(src, dst, bt, dWg0p, dbg0, 1);
        build_dec0<<<buildBlocks, TPB, 0, stream>>>(ybuf, h0, gates, 1, at, 0, zcat, zs, acc);
        cand_seq(src, dst, at, bt, dWc0p, dbc0, h0);

        // layer 1
        build_l1<<<buildBlocks, TPB, 0, stream>>>(h0, h1, nullptr, 0, at, 1, zcat, zs, acc);
        gates_seq(src, dst, bt, dWg1p, dbg1, 0);
        build_l1<<<buildBlocks, TPB, 0, stream>>>(h0, h1, gates, 1, at, 0, zcat, zs, acc);
        cand_seq(src, dst, at, bt, dWc1p, dbc1, h1);

        proj_kernel<<<nodeBlocks, TPB, 0, stream>>>(h1, dWp, dbp, ybuf, out + (size_t)t * NN);
    }
}
